// SimpleNSM_14310831030359
// MI455X (gfx1250) — matrix operand order repaired and checked by reading, not compiled, not run on hardware
//
#include <hip/hip_runtime.h>
#include <hip/hip_bf16.h>
#include <cmath>

// ---------------------------------------------------------------------------
// Problem constants (from the reference)
// ---------------------------------------------------------------------------
#define Bdim    8192
#define IN_DIM  512
#define Ddim    512
#define Nnodes  16
#define OUT_DIM 1024
#define Hheads  4
#define DHead   128   // Ddim / Hheads

// ---------------------------------------------------------------------------
// WMMA types (CDNA5 / gfx1250, wave32)
// ---------------------------------------------------------------------------
typedef __attribute__((ext_vector_type(16))) __bf16 bf16x16;
typedef __attribute__((ext_vector_type(8)))  float  v8f;

__device__ __forceinline__ unsigned short f32_to_bf16_rne(float f) {
    union { float f; unsigned int u; } c; c.f = f;
    unsigned int u = c.u;
    u += 0x7FFFu + ((u >> 16) & 1u);   // round-to-nearest-even
    return (unsigned short)(u >> 16);
}

// ---------------------------------------------------------------------------
// Async global->LDS staging (CDNA5 GLOBAL_LOAD_ASYNC_TO_LDS_B128, ASYNCcnt).
// Builtin signature (upstream): (v4i addrspace(1)*, v4i addrspace(3)*, Ii, Ii)
// Falls back to vector load + ds_store if the builtin is unavailable.
// ---------------------------------------------------------------------------
#if defined(__HIP_DEVICE_COMPILE__) && __has_builtin(__builtin_amdgcn_global_load_async_to_lds_b128)
#define ASYNC_LDS 1
#else
#define ASYNC_LDS 0
#endif

__device__ __forceinline__ void copy16_g2l(const unsigned short* g, unsigned short* l) {
#if ASYNC_LDS
    typedef int v4i32 __attribute__((vector_size(16)));
    typedef __attribute__((address_space(1))) v4i32* gp_t;
    typedef __attribute__((address_space(3))) v4i32* lp_t;
    gp_t gp = (gp_t)(void*)const_cast<unsigned short*>(g);
    lp_t lp = (lp_t)(void*)l;
    __builtin_amdgcn_global_load_async_to_lds_b128(gp, lp, 0, 0);
#else
    *(uint4*)l = *(const uint4*)g;
#endif
}

__device__ __forceinline__ void wait_async_cnt0() {
#if ASYNC_LDS
#if __has_builtin(__builtin_amdgcn_s_wait_asynccnt)
    __builtin_amdgcn_s_wait_asynccnt(0);
#else
    asm volatile("s_wait_asynccnt 0x0" ::: "memory");
#endif
#endif
}

// ---------------------------------------------------------------------------
// bf16 WMMA GEMM:  C[M,Nc](f32) = A[M,K](bf16) @ W[Nc, k_off:k_off+K](bf16)^T (+bias)(+C)
// Optional bf16 mirror of C for feeding the next GEMM without reconversion.
//   - 128x128 tile, BK=64 (two v_wmma_f32_16x16x32_bf16 K-steps per stage)
//   - double-buffered LDS, async global->LDS fills
//   - 256 threads = 8 waves; wave grid 2x4; each wave: 4x2 frags of 16x16
// ---------------------------------------------------------------------------
#define BM 128
#define BN 128
#define BK 64
#define LDS_STRIDE 72   // (BK + 8) shorts -> 144B row stride, 16B aligned

__device__ __forceinline__ void stage_pair(
    unsigned short* __restrict__ AsBuf, unsigned short* __restrict__ WsBuf,
    const unsigned short* __restrict__ A, int lda, int m0, int M,
    const unsigned short* __restrict__ W, int ldw, int k_off, int n0,
    int kb, int tid)
{
    // A tile: 128 rows x 64 cols bf16 = 1024 chunks of 16B (row guarded)
    #pragma unroll
    for (int i = 0; i < 4; ++i) {
        int c   = tid + i * 256;
        int row = c >> 3;
        int col = (c & 7) * 8;
        unsigned short* l = &AsBuf[row * LDS_STRIDE + col];
        if (m0 + row < M) {
            copy16_g2l(A + (size_t)(m0 + row) * lda + kb + col, l);
        } else {
            uint4 z = {0, 0, 0, 0};
            *(uint4*)l = z;
        }
    }
    // W tile: Nc is always a multiple of 128 here, no guard
    #pragma unroll
    for (int i = 0; i < 4; ++i) {
        int c   = tid + i * 256;
        int row = c >> 3;
        int col = (c & 7) * 8;
        copy16_g2l(W + (size_t)(n0 + row) * ldw + k_off + kb + col,
                   &WsBuf[row * LDS_STRIDE + col]);
    }
}

__device__ __forceinline__ void frag_compute(
    const unsigned short* __restrict__ AsBuf, const unsigned short* __restrict__ WsBuf,
    int wrow, int wcol, int l16, int lhalf, v8f (&acc)[4][2])
{
    #pragma unroll
    for (int ks = 0; ks < 2; ++ks) {
        // A 16x32 bf16 frag: lanes 0-15 row M=l: v0..3=K0..7, v4..7=K16..23;
        //                    lanes 16-31 same rows: K8..15 / K24..31.
        bf16x16 af[4];
        #pragma unroll
        for (int mi = 0; mi < 4; ++mi) {
            const unsigned short* base =
                &AsBuf[(wrow + mi * 16 + l16) * LDS_STRIDE + ks * 32 + lhalf * 8];
            union { bf16x16 v; uint4 u[2]; } t;
            t.u[0] = *(const uint4*)(base);
            t.u[1] = *(const uint4*)(base + 16);
            af[mi] = t.v;
        }
        // B 32x16 bf16 frag: lanes 0-15 col N=l hold K0..7,K16..23; lanes 16-31 K8..15,K24..31.
        bf16x16 bf[2];
        #pragma unroll
        for (int ni = 0; ni < 2; ++ni) {
            const unsigned short* base =
                &WsBuf[(wcol + ni * 16 + l16) * LDS_STRIDE + ks * 32 + lhalf * 8];
            union { bf16x16 v; uint4 u[2]; } t;
            t.u[0] = *(const uint4*)(base);
            t.u[1] = *(const uint4*)(base + 16);
            bf[ni] = t.v;
        }
        #pragma unroll
        for (int mi = 0; mi < 4; ++mi)
            #pragma unroll
            for (int ni = 0; ni < 2; ++ni)
                acc[mi][ni] = __builtin_amdgcn_wmma_f32_16x16x32_bf16(
                    false, af[mi], false, bf[ni], (short)0, acc[mi][ni], false, false);
    }
}

template<bool HAS_BIAS, bool ACCUM, bool STORE_BF16>
__global__ void __launch_bounds__(256)
wmma_gemm_bf16(const unsigned short* __restrict__ A, int lda,
               const unsigned short* __restrict__ W, int ldw, int k_off,
               const float* __restrict__ bias,
               float* __restrict__ C, unsigned short* __restrict__ Cbf, int ldc,
               int M, int K)
{
    __shared__ unsigned short Ash[2][BM * LDS_STRIDE];
    __shared__ unsigned short Wsh[2][BN * LDS_STRIDE];

    const int tid   = threadIdx.x;
    const int lane  = tid & 31;
    const int wave  = tid >> 5;       // 0..7
    const int wrow  = (wave >> 2) * 64;
    const int wcol  = (wave & 3) * 32;
    const int m0    = blockIdx.y * BM;
    const int n0    = blockIdx.x * BN;
    const int lhalf = lane >> 4;
    const int l16   = lane & 15;

    v8f acc[4][2];
    #pragma unroll
    for (int mi = 0; mi < 4; ++mi)
        #pragma unroll
        for (int ni = 0; ni < 2; ++ni) { v8f z = {}; acc[mi][ni] = z; }

    // prologue fill of buffer 0
    stage_pair(Ash[0], Wsh[0], A, lda, m0, M, W, ldw, k_off, n0, 0, tid);

    int cur = 0;
    for (int kb = 0; kb < K; kb += BK) {
        wait_async_cnt0();            // current buffer's async fill complete
        __syncthreads();              // ... and visible to all waves
        if (kb + BK < K)              // issue next fill into the other buffer
            stage_pair(Ash[cur ^ 1], Wsh[cur ^ 1], A, lda, m0, M,
                       W, ldw, k_off, n0, kb + BK, tid);
        frag_compute(Ash[cur], Wsh[cur], wrow, wcol, l16, lhalf, acc);
        cur ^= 1;
    }

    // ---- epilogue: C/D layout -> lane l16 = N, VGPR j + 8*lhalf = M ----
    #pragma unroll
    for (int mi = 0; mi < 4; ++mi) {
        #pragma unroll
        for (int ni = 0; ni < 2; ++ni) {
            const int col = n0 + wcol + ni * 16 + l16;
            const float bv = HAS_BIAS ? bias[col] : 0.0f;
            #pragma unroll
            for (int j = 0; j < 8; ++j) {
                const int row = m0 + wrow + mi * 16 + lhalf * 8 + j;
                if (row < M) {
                    size_t idx = (size_t)row * ldc + col;
                    float v = acc[mi][ni][j] + bv;
                    if (ACCUM) v += C[idx];
                    C[idx] = v;
                    if (STORE_BF16) Cbf[idx] = f32_to_bf16_rne(v);
                }
            }
        }
    }
}

// ---------------------------------------------------------------------------
// One-shot fp32 -> bf16 conversion (weights / x / init)
// ---------------------------------------------------------------------------
__global__ void convert_f32_bf16(const float* __restrict__ src,
                                 unsigned short* __restrict__ dst, long long n)
{
    long long i = (long long)blockIdx.x * blockDim.x + threadIdx.x;
    long long stride = (long long)gridDim.x * blockDim.x;
    for (; i < n; i += stride) dst[i] = f32_to_bf16_rne(src[i]);
}

// ---------------------------------------------------------------------------
// Elementwise kernels (grid-stride over B*N*D)
// ---------------------------------------------------------------------------
__device__ __forceinline__ float sigmoidf_(float x) { return 1.0f / (1.0f + expf(-x)); }

// z = sig(AZ[b,d] + CZ[n,d] + bz); Z = z; RSbf = bf16(sig(AR+CR+br) * init[n,d])
__global__ void ew_gates(const float* __restrict__ AZ, const float* __restrict__ AR,
                         const float* __restrict__ CZ, const float* __restrict__ CR,
                         const float* __restrict__ bz, const float* __restrict__ br,
                         const float* __restrict__ init,
                         float* __restrict__ Z, unsigned short* __restrict__ RSbf,
                         long long total)
{
    long long i = (long long)blockIdx.x * blockDim.x + threadIdx.x;
    long long stride = (long long)gridDim.x * blockDim.x;
    for (; i < total; i += stride) {
        int d = (int)(i % Ddim);
        int n = (int)((i / Ddim) % Nnodes);
        long long b = i / ((long long)Ddim * Nnodes);
        float z = sigmoidf_(AZ[b * Ddim + d] + CZ[n * Ddim + d] + bz[d]);
        float r = sigmoidf_(AR[b * Ddim + d] + CR[n * Ddim + d] + br[d]);
        Z[i]    = z;
        RSbf[i] = f32_to_bf16_rne(r * init[n * Ddim + d]);
    }
}

// cand = tanh(HPRE + AH[b,d] + bh); S = (1-Z)*init + Z*cand  (f32 + bf16 mirror)
__global__ void ew_cand_update(const float* __restrict__ HPRE, const float* __restrict__ AH,
                               const float* __restrict__ bh, const float* __restrict__ init,
                               const float* __restrict__ Z,
                               float* __restrict__ S, unsigned short* __restrict__ Sbf,
                               long long total)
{
    long long i = (long long)blockIdx.x * blockDim.x + threadIdx.x;
    long long stride = (long long)gridDim.x * blockDim.x;
    for (; i < total; i += stride) {
        int d = (int)(i % Ddim);
        int n = (int)((i / Ddim) % Nnodes);
        long long b = i / ((long long)Ddim * Nnodes);
        float cand = tanhf(HPRE[i] + AH[b * Ddim + d] + bh[d]);
        float z = Z[i];
        float s = (1.0f - z) * init[n * Ddim + d] + z * cand;
        S[i]   = s;
        Sbf[i] = f32_to_bf16_rne(s);
    }
}

// g = sig(GPRE); S2bf = bf16(g*S + (1-g)*CTX)   (feeds only the final GEMM)
__global__ void ew_final_mix(const float* __restrict__ GPRE, const float* __restrict__ CTX,
                             const float* __restrict__ S, unsigned short* __restrict__ S2bf,
                             long long total)
{
    long long i = (long long)blockIdx.x * blockDim.x + threadIdx.x;
    long long stride = (long long)gridDim.x * blockDim.x;
    for (; i < total; i += stride) {
        float g = sigmoidf_(GPRE[i]);
        S2bf[i] = f32_to_bf16_rne(g * S[i] + (1.0f - g) * CTX[i]);
    }
}

// ---------------------------------------------------------------------------
// Attention over N=16 nodes, per (head, batch) block. ~0.7% of total FLOPs.
// Writes the bf16 mirror directly (only consumer is the Wo GEMM).
// ---------------------------------------------------------------------------
__global__ void __launch_bounds__(128)
attention_kernel(const float* __restrict__ QKV, unsigned short* __restrict__ ATTNbf)
{
    const int h = blockIdx.x;
    const long long b = blockIdx.y;
    const int t = threadIdx.x;

    __shared__ float Qs[Nnodes][DHead];
    __shared__ float Ks[Nnodes][DHead];
    __shared__ float Vs[Nnodes][DHead];
    __shared__ float P[Nnodes][Nnodes];

    const long long rowBase = b * Nnodes * (3 * Ddim);
    #pragma unroll
    for (int i = 0; i < 16; ++i) {
        int idx = t + i * 128;
        int n = idx >> 7, e = idx & 127;
        long long base = rowBase + (long long)n * (3 * Ddim) + h * DHead + e;
        Qs[n][e] = QKV[base];
        Ks[n][e] = QKV[base + Ddim];
        Vs[n][e] = QKV[base + 2 * Ddim];
    }
    __syncthreads();

    #pragma unroll
    for (int i = 0; i < 2; ++i) {
        int idx = t + i * 128;
        int n = idx >> 4, m = idx & 15;
        float s = 0.f;
        #pragma unroll 8
        for (int e = 0; e < DHead; ++e) s += Qs[n][e] * Ks[m][e];
        P[n][m] = s * 0.08838834764831845f;    // 1/sqrt(128)
    }
    __syncthreads();

    if (t < Nnodes) {
        float mx = P[t][0];
        #pragma unroll
        for (int m = 1; m < Nnodes; ++m) mx = fmaxf(mx, P[t][m]);
        float sum = 0.f;
        float e[Nnodes];
        #pragma unroll
        for (int m = 0; m < Nnodes; ++m) { e[m] = expf(P[t][m] - mx); sum += e[m]; }
        float inv = 1.0f / sum;
        #pragma unroll
        for (int m = 0; m < Nnodes; ++m) P[t][m] = e[m] * inv;
    }
    __syncthreads();

    #pragma unroll
    for (int i = 0; i < 16; ++i) {
        int idx = t + i * 128;
        int n = idx >> 7, e = idx & 127;
        float s = 0.f;
        #pragma unroll
        for (int m = 0; m < Nnodes; ++m) s += P[n][m] * Vs[m][e];
        ATTNbf[(b * Nnodes + n) * Ddim + h * DHead + e] = f32_to_bf16_rne(s);
    }
}

// ---------------------------------------------------------------------------
// Host launcher
// ---------------------------------------------------------------------------
static void launch_gemm(const unsigned short* A, int lda,
                        const unsigned short* W, int ldw, int koff,
                        const float* bias, float* C, unsigned short* Cbf, int ldc,
                        int M, int Nc, int K, bool accum, hipStream_t stream)
{
    dim3 grid(Nc / BN, (M + BM - 1) / BM);
    dim3 blk(256);
    if (Cbf) {
        hipLaunchKernelGGL((wmma_gemm_bf16<true,  false, true >), grid, blk, 0, stream, A, lda, W, ldw, koff, bias, C, Cbf, ldc, M, K);
    } else if (accum) {
        hipLaunchKernelGGL((wmma_gemm_bf16<false, true,  false>), grid, blk, 0, stream, A, lda, W, ldw, koff, bias, C, Cbf, ldc, M, K);
    } else if (bias) {
        hipLaunchKernelGGL((wmma_gemm_bf16<true,  false, false>), grid, blk, 0, stream, A, lda, W, ldw, koff, bias, C, Cbf, ldc, M, K);
    } else {
        hipLaunchKernelGGL((wmma_gemm_bf16<false, false, false>), grid, blk, 0, stream, A, lda, W, ldw, koff, bias, C, Cbf, ldc, M, K);
    }
}

static void launch_convert(const float* src, unsigned short* dst, long long n, hipStream_t stream)
{
    int grid = (int)((n + 255) / 256);
    if (grid > 2048) grid = 2048;
    hipLaunchKernelGGL(convert_f32_bf16, dim3(grid), dim3(256), 0, stream, src, dst, n);
}

extern "C" void kernel_launch(void* const* d_in, const int* in_sizes, int n_in,
                              void* d_out, int out_size, void* d_ws, size_t ws_size,
                              hipStream_t stream)
{
    (void)in_sizes; (void)n_in; (void)out_size; (void)ws_size;

    const float* x     = (const float*)d_in[0];
    const float* W_in  = (const float*)d_in[1];
    const float* b_in  = (const float*)d_in[2];
    const float* init  = (const float*)d_in[3];
    const float* Wz    = (const float*)d_in[4];
    const float* bz    = (const float*)d_in[5];
    const float* Wr    = (const float*)d_in[6];
    const float* br    = (const float*)d_in[7];
    const float* Wh    = (const float*)d_in[8];
    const float* bh    = (const float*)d_in[9];
    const float* Wqkv  = (const float*)d_in[10];
    const float* bqkv  = (const float*)d_in[11];
    const float* Wo    = (const float*)d_in[12];
    const float* bo    = (const float*)d_in[13];
    const float* Wg    = (const float*)d_in[14];
    const float* bg    = (const float*)d_in[15];
    const float* W_out = (const float*)d_in[16];
    const float* b_out = (const float*)d_in[17];
    float* out = (float*)d_out;

    // ---- workspace layout ----
    const size_t BD   = (size_t)Bdim * Ddim;            //  4.19 M
    const size_t BND  = (size_t)Bdim * Nnodes * Ddim;   // 67.1  M
    const size_t ND   = (size_t)Nnodes * Ddim;

    float* wsf = (float*)d_ws;
    size_t off = 0;
    float* XP   = wsf + off; off += BD;      // f32 C of gemm1 (bf16 mirror is what's used)
    float* AZ   = wsf + off; off += BD;
    float* AR   = wsf + off; off += BD;
    float* AH   = wsf + off; off += BD;
    float* CZ   = wsf + off; off += ND;
    float* CR   = wsf + off; off += ND;
    float* Zb   = wsf + off; off += BND;     // reused as GPRE after the GRU update
    float* HPRE = wsf + off; off += BND;
    float* Sb   = wsf + off; off += BND;
    float* QKVb = wsf + off; off += BND * 3;
    float* CTXb = wsf + off; off += BND;
    float* GPRE = Zb;

    unsigned short* wsh = (unsigned short*)(wsf + off);
    size_t hoff = 0;
    unsigned short* xbf     = wsh + hoff; hoff += (size_t)Bdim * IN_DIM;
    unsigned short* initbf  = wsh + hoff; hoff += ND;
    unsigned short* Winbf   = wsh + hoff; hoff += (size_t)Ddim * IN_DIM;
    unsigned short* Wzbf    = wsh + hoff; hoff += (size_t)Ddim * 2 * Ddim;
    unsigned short* Wrbf    = wsh + hoff; hoff += (size_t)Ddim * 2 * Ddim;
    unsigned short* Whbf    = wsh + hoff; hoff += (size_t)Ddim * 2 * Ddim;
    unsigned short* Wqkvbf  = wsh + hoff; hoff += (size_t)3 * Ddim * Ddim;
    unsigned short* Wobf    = wsh + hoff; hoff += (size_t)Ddim * Ddim;
    unsigned short* Wgbf    = wsh + hoff; hoff += (size_t)Ddim * 2 * Ddim;
    unsigned short* Woutbf  = wsh + hoff; hoff += (size_t)OUT_DIM * Nnodes * Ddim;
    unsigned short* XPbf    = wsh + hoff; hoff += BD;
    unsigned short* shared3 = wsh + hoff; hoff += BND;   // RSbf -> ATTNbf -> S2bf
    unsigned short* Sbf     = wsh + hoff; hoff += BND;
    unsigned short* CTXbf   = wsh + hoff; hoff += BND;
    unsigned short* RSbf   = shared3;
    unsigned short* ATTNbf = shared3;
    unsigned short* S2bf   = shared3;

    const long long total = (long long)BND;
    const int EW_BLK = 256, EW_GRID = 4096;
    const int BN_M = Bdim * Nnodes;           // 131072 rows

    // 0) one-shot fp32 -> bf16 of all GEMM operands that come from inputs
    launch_convert(x,     xbf,    (long long)Bdim * IN_DIM, stream);
    launch_convert(init,  initbf, (long long)ND, stream);
    launch_convert(W_in,  Winbf,  (long long)Ddim * IN_DIM, stream);
    launch_convert(Wz,    Wzbf,   (long long)Ddim * 2 * Ddim, stream);
    launch_convert(Wr,    Wrbf,   (long long)Ddim * 2 * Ddim, stream);
    launch_convert(Wh,    Whbf,   (long long)Ddim * 2 * Ddim, stream);
    launch_convert(Wqkv,  Wqkvbf, (long long)3 * Ddim * Ddim, stream);
    launch_convert(Wo,    Wobf,   (long long)Ddim * Ddim, stream);
    launch_convert(Wg,    Wgbf,   (long long)Ddim * 2 * Ddim, stream);
    launch_convert(W_out, Woutbf, (long long)OUT_DIM * Nnodes * Ddim, stream);

    // 1) xp = x @ W_in^T + b_in    (bf16 mirror feeds gate GEMMs)
    launch_gemm(xbf, IN_DIM, Winbf, IN_DIM, 0, b_in, XP, XPbf, Ddim, Bdim, Ddim, IN_DIM, false, stream);
    // 2-4) per-batch gate terms: xp @ W*[:, D:2D]^T
    launch_gemm(XPbf, Ddim, Wzbf, 2 * Ddim, Ddim, nullptr, AZ, nullptr, Ddim, Bdim, Ddim, Ddim, false, stream);
    launch_gemm(XPbf, Ddim, Wrbf, 2 * Ddim, Ddim, nullptr, AR, nullptr, Ddim, Bdim, Ddim, Ddim, false, stream);
    launch_gemm(XPbf, Ddim, Whbf, 2 * Ddim, Ddim, nullptr, AH, nullptr, Ddim, Bdim, Ddim, Ddim, false, stream);
    // 5-6) per-node gate terms: init @ W*[:, 0:D]^T  (rank-16)
    launch_gemm(initbf, Ddim, Wzbf, 2 * Ddim, 0, nullptr, CZ, nullptr, Ddim, Nnodes, Ddim, Ddim, false, stream);
    launch_gemm(initbf, Ddim, Wrbf, 2 * Ddim, 0, nullptr, CR, nullptr, Ddim, Nnodes, Ddim, Ddim, false, stream);
    // 7) gates
    hipLaunchKernelGGL(ew_gates, dim3(EW_GRID), dim3(EW_BLK), 0, stream,
                       AZ, AR, CZ, CR, bz, br, init, Zb, RSbf, total);
    // 8) HPRE = (r*s) @ Wh[:, 0:D]^T
    launch_gemm(RSbf, Ddim, Whbf, 2 * Ddim, 0, nullptr, HPRE, nullptr, Ddim, BN_M, Ddim, Ddim, false, stream);
    // 9) cand + GRU update
    hipLaunchKernelGGL(ew_cand_update, dim3(EW_GRID), dim3(EW_BLK), 0, stream,
                       HPRE, AH, bh, init, Zb, Sb, Sbf, total);
    // 10) QKV = S @ Wqkv^T + bqkv
    launch_gemm(Sbf, Ddim, Wqkvbf, Ddim, 0, bqkv, QKVb, nullptr, 3 * Ddim, BN_M, 3 * Ddim, Ddim, false, stream);
    // 11) multihead attention over the 16 nodes
    hipLaunchKernelGGL(attention_kernel, dim3(Hheads, Bdim), dim3(128), 0, stream, QKVb, ATTNbf);
    // 12) CTX = ATTN @ Wo^T + bo   (f32 for the final mix + bf16 for the Wg GEMM)
    launch_gemm(ATTNbf, Ddim, Wobf, Ddim, 0, bo, CTXb, CTXbf, Ddim, BN_M, Ddim, Ddim, false, stream);
    // 13) GPRE = S @ Wg[:, 0:D]^T + bg ; 14) GPRE += CTX @ Wg[:, D:2D]^T
    launch_gemm(Sbf,   Ddim, Wgbf, 2 * Ddim, 0,    bg,      GPRE, nullptr, Ddim, BN_M, Ddim, Ddim, false, stream);
    launch_gemm(CTXbf, Ddim, Wgbf, 2 * Ddim, Ddim, nullptr, GPRE, nullptr, Ddim, BN_M, Ddim, Ddim, true,  stream);
    // 15) S2 = g*S + (1-g)*CTX  (bf16, feeds only the output GEMM)
    hipLaunchKernelGGL(ew_final_mix, dim3(EW_GRID), dim3(EW_BLK), 0, stream, GPRE, CTXb, Sb, S2bf, total);
    // 16) out = S2.reshape(B, N*D) @ W_out^T + b_out
    launch_gemm(S2bf, Nnodes * Ddim, Woutbf, Nnodes * Ddim, 0, b_out, out, nullptr, OUT_DIM,
                Bdim, OUT_DIM, Nnodes * Ddim, false, stream);
}
